// GnnRnnAgent_2800318677701
// MI455X (gfx1250) — compile-verified, matrix operand order checked by reading, not run on hardware
//
#include <hip/hip_runtime.h>

// ---------------------------------------------------------------------------
// Problem constants (match the reference)
// ---------------------------------------------------------------------------
#define NN      262144      // nodes
#define EE      2097152     // edges
#define FF      (EE + NN)   // edges + self loops
#define D_IN    96
#define HH      64          // hidden_dim
#define H2      128         // 2*H
#define ED      5
#define AA      20          // n_actions
#define NEG_SLOPE 0.2f

typedef __attribute__((ext_vector_type(16))) __bf16 v16bf;
typedef __attribute__((ext_vector_type(8)))  float  v8f;

union BFrag {
    unsigned short u[16];
    uint4          q[2];
    v16bf          v;
};

// fp32 -> bf16 bits, round-to-nearest-even (pure integer, no __bf16 arithmetic)
__device__ __forceinline__ unsigned short f2bf(float f) {
    union { float f; unsigned u; } c; c.f = f;
    unsigned u = c.u;
    return (unsigned short)((u + 0x7FFFu + ((u >> 16) & 1u)) >> 16);
}
__device__ __forceinline__ float sigm(float x) { return 1.0f / (1.0f + expf(-x)); }

// ordered-uint mapping so atomicMax(unsigned) == float max
__device__ __forceinline__ unsigned ordf(float f) {
    union { float f; unsigned u; } c; c.f = f;
    return (c.u & 0x80000000u) ? ~c.u : (c.u | 0x80000000u);
}
__device__ __forceinline__ float unordf(unsigned u) {
    union { unsigned u; float f; } c;
    c.u = (u & 0x80000000u) ? (u & 0x7FFFFFFFu) : ~u;
    return c.f;
}

// ---------------------------------------------------------------------------
// Utility kernels
// ---------------------------------------------------------------------------
__global__ void k_zero(float* __restrict__ p, long long n) {
    long long i = (long long)blockIdx.x * blockDim.x + threadIdx.x;
    if (i < n) p[i] = 0.0f;
}

// Pack W[Kin,Kout] (fp32) into WMMA B-fragment-major bf16:
//   dst[((kt*nTiles)+nt)*512 + lane*16 + s]
//   lane: col = nt*16 + (lane&15); kb = (lane>=16)?8:0
//   s<8 : k = kt*32 + kb + s ; s>=8 : k = kt*32 + kb + 16 + (s-8)
// cols >= Kout are zero-filled (padding for Kout=20 -> 32).
__global__ void k_pack_b(const float* __restrict__ W, unsigned short* __restrict__ dst,
                         int Kin, int Kout, int nTiles) {
    int idx = blockIdx.x * blockDim.x + threadIdx.x;
    int total = (Kin >> 5) * nTiles * 512;
    if (idx >= total) return;
    int s    = idx & 15;
    int lane = (idx >> 4) & 31;
    int nt   = (idx >> 9) % nTiles;
    int kt   = idx / (512 * nTiles);
    int kb   = (lane >= 16) ? 8 : 0;
    int k    = kt * 32 + kb + ((s < 8) ? s : (16 + s - 8));
    int col  = nt * 16 + (lane & 15);
    dst[idx] = (col < Kout) ? f2bf(W[k * Kout + col]) : (unsigned short)0;
}

// ---------------------------------------------------------------------------
// Generic WMMA GEMM:  C[N,Kout] = act(A[N,Kin] @ W + bias), W pre-packed bf16
// block = 256 threads (8 waves), 128 rows per block; Kin in {64,96,128}
// ---------------------------------------------------------------------------
__global__ __launch_bounds__(256)
void k_gemm_bf16(const float* __restrict__ A, const unsigned short* __restrict__ Wp,
                 const float* __restrict__ bias, float* __restrict__ C,
                 int Kin, int Kout, int nTiles, int act) {
    __shared__ unsigned short As[128 * 128];   // bf16 activations (32 KB)
    const int tid = threadIdx.x;
    const long long rowBase = (long long)blockIdx.x * 128;

    // stage + convert 128 rows of A into LDS
    const int total = 128 * Kin;
    for (int idx = tid; idx < total; idx += 256) {
        int r = idx / Kin, k = idx - r * Kin;
        As[idx] = f2bf(A[(rowBase + r) * Kin + k]);
    }
    __syncthreads();

    const int wave = tid >> 5;
    const int lane = tid & 31;
    const int m    = lane & 15;
    const int hi   = lane >> 4;          // 0: K 0-7/16-23, 1: K 8-15/24-31
    const int kb   = hi ? 8 : 0;
    const int rowOff = wave * 16;

    for (int nt = 0; nt < nTiles; ++nt) {
        v8f acc = {};
        const int col = nt * 16 + m;
        for (int k0 = 0; k0 < Kin; k0 += 32) {
            BFrag a, b;
            const unsigned short* ar = &As[(rowOff + m) * Kin + k0 + kb];
            a.q[0] = *(const uint4*)(ar);        // K kb..kb+7   (16B aligned: Kin mult of 32... 8-elt=16B)
            a.q[1] = *(const uint4*)(ar + 16);   // K kb+16..kb+23
            const uint4* pb = (const uint4*)(Wp + (((k0 >> 5) * nTiles + nt) << 9) + (lane << 4));
            b.q[0] = pb[0];
            b.q[1] = pb[1];
            acc = __builtin_amdgcn_wmma_f32_16x16x32_bf16(
                      false, a.v, false, b.v, (short)0, acc, false, false);
        }
        if (col < Kout) {
            const float bv = bias ? bias[col] : 0.0f;
#pragma unroll
            for (int r = 0; r < 8; ++r) {
                float v = acc[r] + bv;
                if (act == 1) v = v > 0.0f ? v : 0.0f;
                C[(rowBase + rowOff + r + hi * 8) * Kout + col] = v;
            }
        }
    }
}

// ---------------------------------------------------------------------------
// Edge phase
// ---------------------------------------------------------------------------
__global__ void k_deg_loop(const int* __restrict__ dst, const float* __restrict__ edge_attr,
                           float* __restrict__ deg, float* __restrict__ loop_attr) {
    int e = blockIdx.x * blockDim.x + threadIdx.x;
    if (e >= EE) return;
    int d = dst[e];
    atomicAdd(&deg[d], 1.0f);
#pragma unroll
    for (int k = 0; k < ED; ++k)
        atomicAdd(&loop_attr[(long long)d * ED + k], edge_attr[(long long)e * ED + k]);
}

__global__ void k_loop_fin(float* __restrict__ loop_attr, const float* __restrict__ deg) {
    int i = blockIdx.x * blockDim.x + threadIdx.x;
    if (i >= NN * ED) return;
    loop_attr[i] /= fmaxf(deg[i / ED], 1.0f);
}

// e = leaky_relu(xl[s]+xr[d]+ea@We); logit = e.att ; segment max via atomicMax
__global__ void k_edge_logits(const float* __restrict__ xl, const float* __restrict__ xr,
                              const int* __restrict__ src, const int* __restrict__ dst,
                              const float* __restrict__ edge_attr, const float* __restrict__ loop_attr,
                              const float* __restrict__ We, const float* __restrict__ att,
                              float* __restrict__ logits, unsigned* __restrict__ maxbuf) {
    int f = blockIdx.x * blockDim.x + threadIdx.x;
    if (f >= FF) return;
    int s, d; const float* ea;
    if (f < EE) { s = src[f]; d = dst[f]; ea = edge_attr + (long long)f * ED; }
    else        { s = d = f - EE;         ea = loop_attr + (long long)(f - EE) * ED; }
    const float e0 = ea[0], e1 = ea[1], e2 = ea[2], e3 = ea[3], e4 = ea[4];
    const float4* pl = (const float4*)(xl + (long long)s * HH);
    const float4* pr = (const float4*)(xr + (long long)d * HH);
    const float4* w  = (const float4*)We;          // [ED][16 float4]
    const float4* pa = (const float4*)att;
    float logit = 0.0f;
#pragma unroll 4
    for (int j = 0; j < 16; ++j) {
        float4 a = pl[j], b = pr[j];
        float4 q0 = w[j], q1 = w[16 + j], q2 = w[32 + j], q3 = w[48 + j], q4 = w[64 + j];
        float4 at = pa[j];
        float v;
        v = a.x + b.x + e0*q0.x + e1*q1.x + e2*q2.x + e3*q3.x + e4*q4.x;
        v = v > 0.f ? v : NEG_SLOPE * v; logit += v * at.x;
        v = a.y + b.y + e0*q0.y + e1*q1.y + e2*q2.y + e3*q3.y + e4*q4.y;
        v = v > 0.f ? v : NEG_SLOPE * v; logit += v * at.y;
        v = a.z + b.z + e0*q0.z + e1*q1.z + e2*q2.z + e3*q3.z + e4*q4.z;
        v = v > 0.f ? v : NEG_SLOPE * v; logit += v * at.z;
        v = a.w + b.w + e0*q0.w + e1*q1.w + e2*q2.w + e3*q3.w + e4*q4.w;
        v = v > 0.f ? v : NEG_SLOPE * v; logit += v * at.w;
    }
    logits[f] = logit;
    atomicMax(&maxbuf[d], ordf(logit));
}

__global__ void k_exp_denom(float* __restrict__ logits, const unsigned* __restrict__ maxbuf,
                            const int* __restrict__ dst, float* __restrict__ denom) {
    int f = blockIdx.x * blockDim.x + threadIdx.x;
    if (f >= FF) return;
    int d = (f < EE) ? dst[f] : (f - EE);
    float p = expf(logits[f] - unordf(maxbuf[d]));
    logits[f] = p;                       // overwrite with numerator
    atomicAdd(&denom[d], p);
}

// agg[d] += xl[s] * alpha ; 4 threads per edge, 16 channels each
__global__ void k_aggregate(const float* __restrict__ xl, const int* __restrict__ src,
                            const int* __restrict__ dst, const float* __restrict__ p,
                            const float* __restrict__ denom, float* __restrict__ agg) {
    long long t = (long long)blockIdx.x * blockDim.x + threadIdx.x;
    int f = (int)(t >> 2);
    if (f >= FF) return;
    int q = (int)(t & 3);
    int s, d;
    if (f < EE) { s = src[f]; d = dst[f]; } else { s = d = f - EE; }
    float alpha = p[f] / denom[d];
    const float* xs = xl  + (long long)s * HH + q * 16;
    float*       ag = agg + (long long)d * HH + q * 16;
#pragma unroll
    for (int i = 0; i < 16; ++i) atomicAdd(&ag[i], xs[i] * alpha);
}

__global__ void k_hcat(const float* __restrict__ agg, const float* __restrict__ gat_bias,
                       const float* __restrict__ x, float* __restrict__ h_cat) {
    long long idx = (long long)blockIdx.x * blockDim.x + threadIdx.x;
    if (idx >= (long long)NN * H2) return;
    long long i = idx >> 7; int c = (int)(idx & 127);
    h_cat[idx] = (c < HH) ? fmaxf(agg[i * HH + c] + gat_bias[c], 0.0f)
                          : x[i * HH + (c - HH)];
}

// ---------------------------------------------------------------------------
// Fused GRU: gi = h_cat@W_ih, gh = h_in@W_hh, gates, h -> out_h
// block = 256 (8 waves) handles 16 rows; wave w owns cols [16w,16w+16) of H2
// Wih/Whh pre-packed with nTiles = 24 (384/16)
// ---------------------------------------------------------------------------
__global__ __launch_bounds__(256)
void k_gru(const float* __restrict__ h_cat, const float* __restrict__ h_in,
           const unsigned short* __restrict__ Wih, const unsigned short* __restrict__ Whh,
           const float* __restrict__ bih, const float* __restrict__ bhh,
           float* __restrict__ out_h) {
    __shared__ unsigned short Ac[16 * H2];
    __shared__ unsigned short Ah[16 * H2];
    const int tid = threadIdx.x;
    const long long rowBase = (long long)blockIdx.x * 16;
    for (int idx = tid; idx < 16 * H2; idx += 256) {
        long long g = (rowBase + (idx >> 7)) * H2 + (idx & 127);
        Ac[idx] = f2bf(h_cat[g]);
        Ah[idx] = f2bf(h_in[g]);
    }
    __syncthreads();

    const int wave = tid >> 5, lane = tid & 31;
    const int m = lane & 15, hi = lane >> 4, kb = hi ? 8 : 0;

    v8f gi0 = {}, gi1 = {}, gi2 = {}, gh0 = {}, gh1 = {}, gh2 = {};
    const int c = wave * 16 + m;          // column in [0,128)

#pragma unroll
    for (int k0 = 0; k0 < H2; k0 += 32) {
        BFrag ac, ah;
        const unsigned short* pc = &Ac[m * H2 + k0 + kb];
        const unsigned short* ph = &Ah[m * H2 + k0 + kb];
        ac.q[0] = *(const uint4*)(pc);  ac.q[1] = *(const uint4*)(pc + 16);
        ah.q[0] = *(const uint4*)(ph);  ah.q[1] = *(const uint4*)(ph + 16);
#pragma unroll
        for (int g = 0; g < 3; ++g) {
            const int ntile = wave + g * 8;                 // col tile in [0,24)
            const long long fb = ((long long)((k0 >> 5) * 24 + ntile) << 9) + (lane << 4);
            BFrag bi, bh;
            const uint4* pbi = (const uint4*)(Wih + fb);
            const uint4* pbh = (const uint4*)(Whh + fb);
            bi.q[0] = pbi[0]; bi.q[1] = pbi[1];
            bh.q[0] = pbh[0]; bh.q[1] = pbh[1];
            if (g == 0) {
                gi0 = __builtin_amdgcn_wmma_f32_16x16x32_bf16(false, ac.v, false, bi.v, (short)0, gi0, false, false);
                gh0 = __builtin_amdgcn_wmma_f32_16x16x32_bf16(false, ah.v, false, bh.v, (short)0, gh0, false, false);
            } else if (g == 1) {
                gi1 = __builtin_amdgcn_wmma_f32_16x16x32_bf16(false, ac.v, false, bi.v, (short)0, gi1, false, false);
                gh1 = __builtin_amdgcn_wmma_f32_16x16x32_bf16(false, ah.v, false, bh.v, (short)0, gh1, false, false);
            } else {
                gi2 = __builtin_amdgcn_wmma_f32_16x16x32_bf16(false, ac.v, false, bi.v, (short)0, gi2, false, false);
                gh2 = __builtin_amdgcn_wmma_f32_16x16x32_bf16(false, ah.v, false, bh.v, (short)0, gh2, false, false);
            }
        }
    }

    const float br = bih[c] + bhh[c];
    const float bz = bih[c + 128] + bhh[c + 128];
    const float bin = bih[c + 256], bhn = bhh[c + 256];
#pragma unroll
    for (int r = 0; r < 8; ++r) {
        long long grow = rowBase + r + hi * 8;
        float rg = sigm(gi0[r] + gh0[r] + br);
        float z  = sigm(gi1[r] + gh1[r] + bz);
        float nn = tanhf(gi2[r] + bin + rg * (gh2[r] + bhn));
        float hp = h_in[grow * H2 + c];
        out_h[grow * H2 + c] = (1.0f - z) * nn + z * hp;
    }
}

// ---------------------------------------------------------------------------
// Host: pipeline launch (graph-capture safe: kernels only, scratch in d_ws)
// ---------------------------------------------------------------------------
extern "C" void kernel_launch(void* const* d_in, const int* in_sizes, int n_in,
                              void* d_out, int out_size, void* d_ws, size_t ws_size,
                              hipStream_t stream) {
    const float* inputs   = (const float*)d_in[0];
    const float* hidden   = (const float*)d_in[1];
    const int*   eidx     = (const int*)d_in[2];
    const float* eattr    = (const float*)d_in[3];
    const float* W1 = (const float*)d_in[4],  *b1 = (const float*)d_in[5];
    const float* Wl = (const float*)d_in[6],  *bl = (const float*)d_in[7];
    const float* Wr = (const float*)d_in[8],  *br = (const float*)d_in[9];
    const float* We = (const float*)d_in[10];
    const float* att = (const float*)d_in[11];
    const float* gat_bias = (const float*)d_in[12];
    const float* W_ih = (const float*)d_in[13], *b_ih = (const float*)d_in[14];
    const float* W_hh = (const float*)d_in[15], *b_hh = (const float*)d_in[16];
    const float* W2 = (const float*)d_in[17],  *b2 = (const float*)d_in[18];

    const int* src = eidx;
    const int* dst = eidx + EE;

    float* q_out = (float*)d_out;                      // [N, 20]
    float* h_out = q_out + (size_t)NN * AA;            // [N, 128]

    // ---- workspace carve-up (256B aligned; zero-region is contiguous) ----
    char* ws = (char*)d_ws;  size_t off = 0;
    auto carve = [&](size_t bytes) {
        void* p = ws + off;
        off = (off + bytes + 255) & ~(size_t)255;
        return p;
    };
    float* x     = (float*)carve((size_t)NN * HH * 4);   // relu(fc1)
    float* xl    = (float*)carve((size_t)NN * HH * 4);
    float* xr    = (float*)carve((size_t)NN * HH * 4);
    float* hcat  = (float*)carve((size_t)NN * H2 * 4);
    float* logit = (float*)carve((size_t)FF * 4);        // later holds p
    // zero-region: agg | deg | loop_attr | denom | maxbuf  (72*N floats, contiguous)
    float*    agg      = (float*)carve((size_t)NN * HH * 4);
    float*    deg      = (float*)carve((size_t)NN * 4);
    float*    loop_at  = (float*)carve((size_t)NN * ED * 4);
    float*    denom    = (float*)carve((size_t)NN * 4);
    unsigned* maxbuf   = (unsigned*)carve((size_t)NN * 4);
    // packed bf16 weights: (Kin/32)*nTiles*512 shorts each
    unsigned short* W1p  = (unsigned short*)carve((size_t)3 * 4 * 512 * 2);   // 96x64
    unsigned short* Wlp  = (unsigned short*)carve((size_t)2 * 4 * 512 * 2);   // 64x64
    unsigned short* Wrp  = (unsigned short*)carve((size_t)2 * 4 * 512 * 2);   // 64x64
    unsigned short* Wihp = (unsigned short*)carve((size_t)4 * 24 * 512 * 2);  // 128x384
    unsigned short* Whhp = (unsigned short*)carve((size_t)4 * 24 * 512 * 2);  // 128x384
    unsigned short* W2p  = (unsigned short*)carve((size_t)4 * 2 * 512 * 2);   // 128x20->32

    const int T = 256;

    // 0) zero accumulators (agg..maxbuf contiguous: 72*N floats)
    {
        long long nz = (long long)NN * 72;
        k_zero<<<dim3((unsigned)((nz + T - 1) / T)), dim3(T), 0, stream>>>(agg, nz);
    }
    // 1) pack weights into WMMA B-fragment layout (bf16)
    k_pack_b<<<dim3((3*4*512)/T),  dim3(T), 0, stream>>>(W1,   W1p,  D_IN, HH,  4);
    k_pack_b<<<dim3((2*4*512)/T),  dim3(T), 0, stream>>>(Wl,   Wlp,  HH,   HH,  4);
    k_pack_b<<<dim3((2*4*512)/T),  dim3(T), 0, stream>>>(Wr,   Wrp,  HH,   HH,  4);
    k_pack_b<<<dim3((4*24*512)/T), dim3(T), 0, stream>>>(W_ih, Wihp, H2,   384, 24);
    k_pack_b<<<dim3((4*24*512)/T), dim3(T), 0, stream>>>(W_hh, Whhp, H2,   384, 24);
    k_pack_b<<<dim3((4*2*512)/T),  dim3(T), 0, stream>>>(W2,   W2p,  H2,   AA,  2);

    // 2) fc1: x = relu(inputs @ W1 + b1)            [N,96]x[96,64]
    k_gemm_bf16<<<dim3(NN/128), dim3(T), 0, stream>>>(inputs, W1p, b1, x, D_IN, HH, 4, 1);
    // 3) GAT linear transforms
    k_gemm_bf16<<<dim3(NN/128), dim3(T), 0, stream>>>(x, Wlp, bl, xl, HH, HH, 4, 0);
    k_gemm_bf16<<<dim3(NN/128), dim3(T), 0, stream>>>(x, Wrp, br, xr, HH, HH, 4, 0);

    // 4) degrees + mean self-loop attrs
    k_deg_loop<<<dim3(EE/T), dim3(T), 0, stream>>>(dst, eattr, deg, loop_at);
    k_loop_fin<<<dim3((NN*ED + T-1)/T), dim3(T), 0, stream>>>(loop_at, deg);

    // 5) edge scores + segment max
    k_edge_logits<<<dim3((FF + T-1)/T), dim3(T), 0, stream>>>(
        xl, xr, src, dst, eattr, loop_at, We, att, logit, maxbuf);
    // 6) softmax numerators + denominators
    k_exp_denom<<<dim3((FF + T-1)/T), dim3(T), 0, stream>>>(logit, maxbuf, dst, denom);
    // 7) weighted scatter-add aggregation
    {
        long long nt = (long long)FF * 4;
        k_aggregate<<<dim3((unsigned)((nt + T-1)/T)), dim3(T), 0, stream>>>(
            xl, src, dst, logit, denom, agg);
    }
    // 8) h_gat = relu(agg + bias); h_cat = [h_gat, x]
    k_hcat<<<dim3((unsigned)(((long long)NN*H2 + T-1)/T)), dim3(T), 0, stream>>>(
        agg, gat_bias, x, hcat);

    // 9) fused GRU -> h_out
    k_gru<<<dim3(NN/16), dim3(T), 0, stream>>>(hcat, hidden, Wihp, Whhp, b_ih, b_hh, h_out);

    // 10) fc2: q = h @ W2 + b2                      [N,128]x[128,20] (pad to 32)
    k_gemm_bf16<<<dim3(NN/128), dim3(T), 0, stream>>>(h_out, W2p, b2, q_out, H2, AA, 2, 0);
}
